// _FusedGriffinBlockReplacement_9354438770832
// MI455X (gfx1250) — compile-verified, matrix operand order checked by reading, not run on hardware
//
#include <hip/hip_runtime.h>

#define DEV __device__ __forceinline__

typedef __attribute__((ext_vector_type(16))) __bf16 v16bf;
typedef __attribute__((ext_vector_type(8)))  float  v8f;

// Problem constants (reference: B=4, S=2048, D=1024, K=4 taps, F=4096)
constexpr int B_ = 4, S_ = 2048, D_ = 1024, KT_ = 4, F_ = 4096;
constexpr int M_ = B_ * S_;   // 8192 token rows

// ---------- small math helpers ----------
DEV unsigned short f2bf(float f) {             // round-to-nearest-even f32->bf16
  unsigned u = __float_as_uint(f);
  unsigned r = 0x7FFFu + ((u >> 16) & 1u);
  return (unsigned short)((u + r) >> 16);
}
DEV float sigmoidf_(float x) { return __fdividef(1.0f, 1.0f + __expf(-x)); }
DEV float geluf_(float x) {    // tanh approximation (jax.nn.gelu default)
  float x3 = x * x * x;
  float t = tanhf(0.7978845608028654f * (x + 0.044715f * x3));
  return 0.5f * x * (1.0f + t);
}

// ---------- weight convert: f32 W[K][N] -> bf16 Wt[N][K] ----------
__global__ void wt_convert_kernel(const float* __restrict__ W,
                                  unsigned short* __restrict__ Wt,
                                  int K, int N) {
  int idx = blockIdx.x * blockDim.x + threadIdx.x;
  if (idx >= K * N) return;
  int n = idx / K;
  int k = idx - n * K;
  Wt[(size_t)n * K + k] = f2bf(W[(size_t)k * N + n]);
}

// ---------- RMSNorm: one block (256 thr) per token row of D=1024 ----------
__global__ __launch_bounds__(256)
void rmsnorm_kernel(const float* __restrict__ x, const float* __restrict__ w,
                    float* __restrict__ outF, unsigned short* __restrict__ outH) {
  int row = blockIdx.x;
  const float* xr = x + (size_t)row * D_;
  __shared__ float red[8];
  float4 v = ((const float4*)xr)[threadIdx.x];          // D/4 == blockDim
  float ss = v.x * v.x + v.y * v.y + v.z * v.z + v.w * v.w;
  for (int off = 16; off > 0; off >>= 1) ss += __shfl_down(ss, off, 32);
  if ((threadIdx.x & 31) == 0) red[threadIdx.x >> 5] = ss;
  __syncthreads();
  if (threadIdx.x < 8) {
    float t = red[threadIdx.x];
    for (int off = 4; off > 0; off >>= 1) t += __shfl_down(t, off, 8);
    if (threadIdx.x == 0) red[0] = t;
  }
  __syncthreads();
  float rms = rsqrtf(red[0] * (1.0f / (float)D_) + 1e-6f);
  int d0 = threadIdx.x * 4;
  float o0 = v.x * rms * w[d0 + 0];
  float o1 = v.y * rms * w[d0 + 1];
  float o2 = v.z * rms * w[d0 + 2];
  float o3 = v.w * rms * w[d0 + 3];
  if (outF) ((float4*)(outF + (size_t)row * D_))[threadIdx.x] = make_float4(o0, o1, o2, o3);
  if (outH) {
    unsigned short* h = outH + (size_t)row * D_ + d0;
    h[0] = f2bf(o0); h[1] = f2bf(o1); h[2] = f2bf(o2); h[3] = f2bf(o3);
  }
}

// ---------- causal depthwise conv -> bf16 into left half of concat ----------
__global__ __launch_bounds__(256)
void dwconv_kernel(const float* __restrict__ xn, const float* __restrict__ kern,
                   const float* __restrict__ bias, unsigned short* __restrict__ cat) {
  int idx = blockIdx.x * blockDim.x + threadIdx.x;      // over M_*D_
  int d = idx & (D_ - 1);
  int row = idx >> 10;                                  // b*S + t
  int t = row & (S_ - 1);
  float acc = bias[d];
#pragma unroll
  for (int k = 0; k < KT_; ++k)
    if (t - k >= 0) acc += kern[d * KT_ + k] * xn[(size_t)(row - k) * D_ + d];
  cat[(size_t)row * (2 * D_) + d] = f2bf(acc);
}

// ---------- RG-LRU sequential scan: one thread per (b,d) channel ----------
__global__ __launch_bounds__(256)
void scan_kernel(const float* __restrict__ a_, const float* __restrict__ g_,
                 const float* __restrict__ xn, unsigned short* __restrict__ cat) {
  int ch = blockIdx.x * blockDim.x + threadIdx.x;       // 0..B*D-1
  int b = ch >> 10, d = ch & (D_ - 1);
  float h = 0.f;
  size_t base = (size_t)b * S_ * D_ + d;
  for (int t = 0; t < S_; ++t) {
    size_t i = base + (size_t)t * D_;
    float a = a_[i], g = g_[i], x = xn[i];
    float u = sqrtf(fmaxf(1.f - a * a, 0.f)) * g * x;
    h = a * h + u;
    cat[(size_t)(b * S_ + t) * (2 * D_) + D_ + d] = f2bf(h);
  }
}

// ---------- bf16 WMMA GEMM with fused epilogues ----------
// A: bf16 [M][K] row-major, Bt: bf16 [N][K] (W transposed), f32 accumulate.
// Block 256 thr = 8 waves arranged 4(M)x2(N); wave tile 32(M)x64(N)
// = 2 A-frags x 4 B-frags = 8 v_wmma per 32-wide K-step.
// Block tile 128x128; grid dim3(N/128, M/128).
// EPI: 0 sigmoid->outF ; 1 a=exp(-8*softplus(e0[n])*sigmoid)->outF ;
//      2 mixer: vel=sig(e3[n])*e1[o]+(acc+e2[n]) ->outF, xnew=e0[o]+vel ->outF2 ;
//      3 gelu(acc+e0[n]) -> bf16 outH ; 4 outF[o]=e0[o]+acc+e1[n]
template <int EPI>
__global__ __launch_bounds__(256)
void gemm_bf16(const unsigned short* __restrict__ A,
               const unsigned short* __restrict__ Bt,
               int M, int N, int K,
               float* __restrict__ outF, float* __restrict__ outF2,
               unsigned short* __restrict__ outH,
               const float* __restrict__ e0, const float* __restrict__ e1,
               const float* __restrict__ e2, const float* __restrict__ e3) {
  const int lane = threadIdx.x & 31;
  const int wid  = threadIdx.x >> 5;
  const int wm = wid >> 1, wn = wid & 1;              // 4 x 2 waves
  const int m0 = blockIdx.y * 128 + wm * 32;
  const int n0 = blockIdx.x * 128 + wn * 64;
  const int r  = lane & 15;
  const int hh = lane >> 4;      // lane half selects K sub-blocks (ISA 7.12.2)

  v8f c[2][4] = {};
  const unsigned short* ap[2];
  const unsigned short* bp[4];
#pragma unroll
  for (int i = 0; i < 2; ++i)
    ap[i] = A + (size_t)(m0 + 16 * i + r) * K + hh * 8;
#pragma unroll
  for (int j = 0; j < 4; ++j)
    bp[j] = Bt + (size_t)(n0 + 16 * j + r) * K + hh * 8;

  for (int k0 = 0; k0 < K; k0 += 32) {
    __builtin_prefetch(ap[0] + k0 + 64, 0, 3);   // K-tile two steps ahead, near-cache policy
    __builtin_prefetch(bp[0] + k0 + 64, 0, 3);
    v16bf a[2], b[4];
#pragma unroll
    for (int i = 0; i < 2; ++i) {
      union { v16bf v; unsigned u[8]; } f;
#pragma unroll
      for (int p = 0; p < 4; ++p) {
        f.u[p]     = *(const unsigned*)(ap[i] + k0 + 2 * p);       // K 0..7 (+half*8)
        f.u[4 + p] = *(const unsigned*)(ap[i] + k0 + 16 + 2 * p);  // K 16..23 (+half*8)
      }
      a[i] = f.v;
    }
#pragma unroll
    for (int j = 0; j < 4; ++j) {
      union { v16bf v; unsigned u[8]; } f;
#pragma unroll
      for (int p = 0; p < 4; ++p) {
        f.u[p]     = *(const unsigned*)(bp[j] + k0 + 2 * p);
        f.u[4 + p] = *(const unsigned*)(bp[j] + k0 + 16 + 2 * p);
      }
      b[j] = f.v;
    }
#pragma unroll
    for (int i = 0; i < 2; ++i)
#pragma unroll
      for (int j = 0; j < 4; ++j)
        c[i][j] = __builtin_amdgcn_wmma_f32_16x16x32_bf16(
            false, a[i], false, b[j], (short)0, c[i][j], false, false);
  }

#pragma unroll
  for (int i = 0; i < 2; ++i) {
#pragma unroll
    for (int j = 0; j < 4; ++j) {
      int n = n0 + 16 * j + r;
#pragma unroll
      for (int v = 0; v < 8; ++v) {
        int m = m0 + 16 * i + v + 8 * hh;     // D-layout: M = vgpr + 8*(lane>=16)
        size_t o = (size_t)m * N + n;
        float acc = c[i][j][v];
        if constexpr (EPI == 0) {
          outF[o] = sigmoidf_(acc);
        } else if constexpr (EPI == 1) {
          float lam = e0[n];
          float sp = (lam > 20.f) ? lam : log1pf(__expf(lam));
          outF[o] = __expf(-8.0f * sp * sigmoidf_(acc));
        } else if constexpr (EPI == 2) {
          float mix  = acc + e2[n];
          float beta = sigmoidf_(e3[n]);
          float vel  = beta * e1[o] + mix;
          outF[o]  = vel;             // velocity -> d_out (second output)
          outF2[o] = e0[o] + vel;     // x_new
        } else if constexpr (EPI == 3) {
          outH[o] = f2bf(geluf_(acc + e0[n]));
        } else {                      // EPI == 4
          outF[o] = e0[o] + acc + e1[n];
        }
      }
    }
  }
}

// ---------- workspace layout ----------
constexpr size_t SZ_XN_F = (size_t)M_ * D_ * 4;
constexpr size_t SZ_XN_H = (size_t)M_ * D_ * 2;
constexpr size_t SZ_GATE = SZ_XN_F;
constexpr size_t SZ_A    = SZ_XN_F;
constexpr size_t SZ_CAT  = (size_t)M_ * 2 * D_ * 2;
constexpr size_t SZ_XNEW = SZ_XN_F;
constexpr size_t SZ_NORM = SZ_XN_H;
constexpr size_t SZ_H1   = (size_t)M_ * F_ * 2;
constexpr size_t SZ_WG   = (size_t)D_ * D_ * 2;
constexpr size_t SZ_WO   = (size_t)D_ * 2 * D_ * 2;
constexpr size_t SZ_W1   = (size_t)D_ * F_ * 2;

extern "C" void kernel_launch(void* const* d_in, const int* in_sizes, int n_in,
                              void* d_out, int out_size, void* d_ws, size_t ws_size,
                              hipStream_t stream) {
  const float* x          = (const float*)d_in[0];
  const float* velocity   = (const float*)d_in[1];
  const float* pre_norm_w = (const float*)d_in[2];
  const float* conv_k     = (const float*)d_in[3];
  const float* conv_b     = (const float*)d_in[4];
  const float* W_gate     = (const float*)d_in[5];
  const float* W_a        = (const float*)d_in[6];
  const float* lam        = (const float*)d_in[7];
  const float* W_out      = (const float*)d_in[8];
  const float* b_out      = (const float*)d_in[9];
  const float* log_beta   = (const float*)d_in[10];
  const float* ffn_norm_w = (const float*)d_in[11];
  const float* W_ff1      = (const float*)d_in[12];
  const float* b_ff1      = (const float*)d_in[13];
  const float* W_ff2      = (const float*)d_in[14];
  const float* b_ff2      = (const float*)d_in[15];

  float* out1  = (float*)d_out;                    // x + ffn
  float* out_v = (float*)d_out + (size_t)M_ * D_;  // velocity

  char* w = (char*)d_ws;
  float*          xn_f   = (float*)w;          w += SZ_XN_F;
  unsigned short* xn_h   = (unsigned short*)w; w += SZ_XN_H;
  float*          gate_f = (float*)w;          w += SZ_GATE;
  float*          a_f    = (float*)w;          w += SZ_A;
  unsigned short* cat    = (unsigned short*)w; w += SZ_CAT;
  float*          xnew   = (float*)w;          w += SZ_XNEW;
  unsigned short* normed = (unsigned short*)w; w += SZ_NORM;
  unsigned short* h1     = (unsigned short*)w; w += SZ_H1;
  unsigned short* wgT    = (unsigned short*)w; w += SZ_WG;
  unsigned short* waT    = (unsigned short*)w; w += SZ_WG;
  unsigned short* woT    = (unsigned short*)w; w += SZ_WO;
  unsigned short* w1T    = (unsigned short*)w; w += SZ_W1;
  unsigned short* w2T    = (unsigned short*)w; w += SZ_W1;

  // 1. weights -> bf16 transposed [N][K]
  wt_convert_kernel<<<(D_ * D_ + 255) / 256, 256, 0, stream>>>(W_gate, wgT, D_, D_);
  wt_convert_kernel<<<(D_ * D_ + 255) / 256, 256, 0, stream>>>(W_a,   waT, D_, D_);
  wt_convert_kernel<<<(2 * D_ * D_ + 255) / 256, 256, 0, stream>>>(W_out, woT, 2 * D_, D_);
  wt_convert_kernel<<<(D_ * F_ + 255) / 256, 256, 0, stream>>>(W_ff1, w1T, D_, F_);
  wt_convert_kernel<<<(F_ * D_ + 255) / 256, 256, 0, stream>>>(W_ff2, w2T, F_, D_);

  // 2. RMSNorm(x) -> f32 + bf16
  rmsnorm_kernel<<<M_, 256, 0, stream>>>(x, pre_norm_w, xn_f, xn_h);

  // 3. depthwise causal conv -> cat[:, 0:D] (bf16)
  dwconv_kernel<<<(M_ * D_) / 256, 256, 0, stream>>>(xn_f, conv_k, conv_b, cat);

  // 4. gate & a logits (WMMA bf16) with fused activations
  dim3 gD(D_ / 128, M_ / 128);
  gemm_bf16<0><<<gD, 256, 0, stream>>>(xn_h, wgT, M_, D_, D_,
      gate_f, nullptr, nullptr, nullptr, nullptr, nullptr, nullptr);
  gemm_bf16<1><<<gD, 256, 0, stream>>>(xn_h, waT, M_, D_, D_,
      a_f, nullptr, nullptr, lam, nullptr, nullptr, nullptr);

  // 5. RG-LRU scan -> cat[:, D:2D] (bf16)
  scan_kernel<<<(B_ * D_) / 256, 256, 0, stream>>>(a_f, gate_f, xn_f, cat);

  // 6. W_out GEMM + velocity/residual epilogue
  gemm_bf16<2><<<gD, 256, 0, stream>>>(cat, woT, M_, D_, 2 * D_,
      out_v, xnew, nullptr, x, velocity, b_out, log_beta);

  // 7. RMSNorm(x_new) -> bf16
  rmsnorm_kernel<<<M_, 256, 0, stream>>>(xnew, ffn_norm_w, nullptr, normed);

  // 8. FFN up + GELU (bf16 out)
  dim3 g1(F_ / 128, M_ / 128);
  gemm_bf16<3><<<g1, 256, 0, stream>>>(normed, w1T, M_, F_, D_,
      nullptr, nullptr, h1, b_ff1, nullptr, nullptr, nullptr);

  // 9. FFN down + bias + residual -> d_out first output
  gemm_bf16<4><<<gD, 256, 0, stream>>>(h1, w2T, M_, D_, F_,
      out1, nullptr, nullptr, xnew, b_ff2, nullptr, nullptr);
}
